// NMT_72327249264995
// MI455X (gfx1250) — matrix operand order repaired and checked by reading, not compiled, not run on hardware
//
#include <hip/hip_runtime.h>
#include <hip/hip_bf16.h>

// ---------------------------------------------------------------------------
// NMT forward (CNN + biLSTM encoder + attention decoder + vocab log-softmax)
// for MI455X / gfx1250.  All large contractions run on v_wmma_f32_16x16x32_bf16
// (bf16 A/B, fp32 accumulate).  Weights are converted to bf16 once per call;
// GEMM inner loops are pure b128 loads + WMMA.  LSTM steps fuse the gate
// nonlinearity into the GEMM epilogue (M32xN16 wave tile x 4 gate blocks),
// with ping-pong h state to avoid intra-kernel RAW hazards.
// ---------------------------------------------------------------------------

typedef __attribute__((ext_vector_type(16))) __bf16 v16bf;
typedef __attribute__((ext_vector_type(8)))  __bf16 v8bf;
typedef __attribute__((ext_vector_type(8)))  float  v8f;

#define NB 64      // batch
#define NS 48      // src len
#define NT 48      // tgt len
#define NE 512     // embed dim
#define NH 1024    // hidden
#define NV 32000   // vocab

// Segmented GEMM operands:  C[M,N] = A * B^T
//   A row m, col k:  concat of up to 3 bf16 buffers (row stride == segment len)
//   B row n, col k:  concat of up to 2 bf16 weight buffers (row-major (N,K))
// K-tiles (32) never straddle segment boundaries (all lens multiples of 512).
struct GemmArgs {
  const __bf16* a0; int l0;
  const __bf16* a1; int l1;
  const __bf16* a2; int l2;
  const __bf16* b0; int bl0;
  const __bf16* b1; int bl1;
  int M, N, K;
};

__device__ __forceinline__ const __bf16* a_seg(const GemmArgs& g, int k, int& len, int& ko) {
  if (k < g.l0) { len = g.l0; ko = k; return g.a0; }
  k -= g.l0;
  if (k < g.l1) { len = g.l1; ko = k; return g.a1; }
  k -= g.l1; len = g.l2; ko = k; return g.a2;
}
__device__ __forceinline__ const __bf16* b_seg(const GemmArgs& g, int k, int& len, int& ko) {
  if (k < g.bl0) { len = g.bl0; ko = k; return g.b0; }
  k -= g.bl0; len = g.bl1; ko = k; return g.b1;
}

// Fragment layouts per CDNA5 ISA 7.12.2 (wave32):
//   A 16x32 bf16 : lanes 0-15 row M=l, halves {K0..7, K16..23};
//                  lanes 16-31 row M=l-16, halves {K8..15, K24..31}
//   B 32x16 bf16 : lane n=l%16, same K split as A: lanes<16 K0..7,16..23; lanes>=16 K8..15,24..31
//   C 16x16 f32  : VGPR r -> (M=r,N=lane) lanes<16 ; (M=8+r,N=lane-16) lanes>=16
union AFrag { v16bf v; v8bf h[2]; };

// 32(M) x 64(N) wave tile: all 12 b128 loads issued before the 8 WMMAs so the
// scheduler can stagger s_wait_loadcnt instead of waiting to zero per fragment.
__device__ __forceinline__ void gemm_mainloop(const GemmArgs& g, int tileM, int tileN,
                                              v8f acc[2][4]) {
  const int lane = threadIdx.x & 31;
  const int half = lane >> 4;
  const int l15  = lane & 15;
  for (int k0 = 0; k0 < g.K; k0 += 32) {
    int alen, ako;
    const __bf16* ap = a_seg(g, k0, alen, ako);
    AFrag A[2];
#pragma unroll
    for (int mi = 0; mi < 2; ++mi) {
      const __bf16* arow = ap + (size_t)(tileM + mi * 16 + l15) * alen + ako + half * 8;
      A[mi].h[0] = *(const v8bf*)(arow);
      A[mi].h[1] = *(const v8bf*)(arow + 16);
    }
    if (k0 + 32 < g.K) {           // gfx1250 global_prefetch_b8 of next A tile
      int al2, ak2;
      const __bf16* ap2 = a_seg(g, k0 + 32, al2, ak2);
      __builtin_prefetch(ap2 + (size_t)(tileM + l15) * al2 + ak2 + half * 8, 0, 0);
    }
    int blen, bko;
    const __bf16* bp = b_seg(g, k0, blen, bko);
    AFrag B[4];
#pragma unroll
    for (int j = 0; j < 4; ++j) {
      const __bf16* brow = bp + (size_t)(tileN + j * 16 + l15) * blen + bko + half * 8;
      B[j].h[0] = *(const v8bf*)(brow);
      B[j].h[1] = *(const v8bf*)(brow + 16);
    }
#pragma unroll
    for (int j = 0; j < 4; ++j) {
      acc[0][j] = __builtin_amdgcn_wmma_f32_16x16x32_bf16(
          false, A[0].v, false, B[j].v, (short)0, acc[0][j], false, false);
      acc[1][j] = __builtin_amdgcn_wmma_f32_16x16x32_bf16(
          false, A[1].v, false, B[j].v, (short)0, acc[1][j], false, false);
    }
  }
}

// 32(M) x 16(N) variant: B row base given explicitly (used per gate block).
__device__ __forceinline__ void gemm_mainloop_n16(const GemmArgs& g, int tileM, int nBase,
                                                  v8f acc[2]) {
  const int lane = threadIdx.x & 31;
  const int half = lane >> 4;
  const int l15  = lane & 15;
  for (int k0 = 0; k0 < g.K; k0 += 32) {
    int alen, ako;
    const __bf16* ap = a_seg(g, k0, alen, ako);
    AFrag A[2];
#pragma unroll
    for (int mi = 0; mi < 2; ++mi) {
      const __bf16* arow = ap + (size_t)(tileM + mi * 16 + l15) * alen + ako + half * 8;
      A[mi].h[0] = *(const v8bf*)(arow);
      A[mi].h[1] = *(const v8bf*)(arow + 16);
    }
    int blen, bko;
    const __bf16* bp = b_seg(g, k0, blen, bko);
    const __bf16* brow = bp + (size_t)(nBase + l15) * blen + bko + half * 8;
    AFrag B;
    B.h[0] = *(const v8bf*)(brow);
    B.h[1] = *(const v8bf*)(brow + 16);
    acc[0] = __builtin_amdgcn_wmma_f32_16x16x32_bf16(
        false, A[0].v, false, B.v, (short)0, acc[0], false, false);
    acc[1] = __builtin_amdgcn_wmma_f32_16x16x32_bf16(
        false, A[1].v, false, B.v, (short)0, acc[1], false, false);
  }
}

__device__ __forceinline__ float sigmf(float x) { return 1.f / (1.f + expf(-x)); }

// Generic GEMM + bias(+bias2) + optional tanh, dual-precision stores.
__global__ void gemm_bias_act(GemmArgs g, const float* bias1, const float* bias2,
                              float* outF, __bf16* outB, int act) {
  const int wave   = blockIdx.x * (blockDim.x >> 5) + (threadIdx.x >> 5);
  const int tilesM = g.M >> 5;
  const int tilesN = g.N >> 6;
  if (wave >= tilesM * tilesN) return;   // whole-wave uniform
  const int tileM = (wave % tilesM) << 5;
  const int tileN = (wave / tilesM) << 6;

  v8f zero = {0.f, 0.f, 0.f, 0.f, 0.f, 0.f, 0.f, 0.f};
  v8f acc[2][4] = {{zero, zero, zero, zero}, {zero, zero, zero, zero}};
  gemm_mainloop(g, tileM, tileN, acc);

  const int lane = threadIdx.x & 31;
  const int half = lane >> 4;
  const int l15  = lane & 15;
#pragma unroll
  for (int mi = 0; mi < 2; ++mi) {
#pragma unroll
    for (int j = 0; j < 4; ++j) {
      const int n = tileN + j * 16 + l15;
      float bsum = (bias1 ? bias1[n] : 0.f) + (bias2 ? bias2[n] : 0.f);
      union { v8f v; float e[8]; } C; C.v = acc[mi][j];
#pragma unroll
      for (int r = 0; r < 8; ++r) {
        const int m = tileM + mi * 16 + r + 8 * half;
        float v = C.e[r] + bsum;
        if (act) v = tanhf(v);
        if (outF) outF[(size_t)m * g.N + n] = v;
        if (outB) outB[(size_t)m * g.N + n] = (__bf16)v;
      }
    }
  }
}

// Fused LSTM step: gates GEMM (N=4H via 4 gate-block passes) + nonlinearity.
// h is ping-pong buffered by the caller (reads h_in via g, writes h_out);
// c is updated in place (each (m,n) owned by exactly one wave).
__global__ void lstm_step_fused(GemmArgs g, const float* bias1, const float* bias2,
                                float* c32, __bf16* h_out, __bf16* c_bf,
                                float* h_f32, __bf16* h_copy, int copy_stride) {
  const int wave   = blockIdx.x * (blockDim.x >> 5) + (threadIdx.x >> 5);
  const int tilesM = g.M >> 5;          // M = 64 -> 2
  const int tilesN = NH >> 4;           // 64 n-tiles of 16
  if (wave >= tilesM * tilesN) return;
  const int tileM = (wave % tilesM) << 5;
  const int tileN = (wave / tilesM) << 4;

  v8f zero = {0.f, 0.f, 0.f, 0.f, 0.f, 0.f, 0.f, 0.f};
  v8f acc[4][2];
#pragma unroll
  for (int gt = 0; gt < 4; ++gt) {
    acc[gt][0] = zero; acc[gt][1] = zero;
    gemm_mainloop_n16(g, tileM, gt * NH + tileN, acc[gt]);
  }

  const int lane = threadIdx.x & 31;
  const int half = lane >> 4;
  const int l15  = lane & 15;
  const int n = tileN + l15;
  const float bi = (bias1 ? bias1[n] : 0.f)          + (bias2 ? bias2[n] : 0.f);
  const float bf = (bias1 ? bias1[NH + n] : 0.f)     + (bias2 ? bias2[NH + n] : 0.f);
  const float bg = (bias1 ? bias1[2 * NH + n] : 0.f) + (bias2 ? bias2[2 * NH + n] : 0.f);
  const float bo = (bias1 ? bias1[3 * NH + n] : 0.f) + (bias2 ? bias2[3 * NH + n] : 0.f);
#pragma unroll
  for (int mi = 0; mi < 2; ++mi) {
    union { v8f v; float e[8]; } Ci, Cf, Cg, Co;
    Ci.v = acc[0][mi]; Cf.v = acc[1][mi]; Cg.v = acc[2][mi]; Co.v = acc[3][mi];
#pragma unroll
    for (int r = 0; r < 8; ++r) {
      const int m = tileM + mi * 16 + r + 8 * half;
      const size_t idx = (size_t)m * NH + n;
      const float iv = Ci.e[r] + bi, fv = Cf.e[r] + bf;
      const float gv = Cg.e[r] + bg, ov = Co.e[r] + bo;
      const float cn = sigmf(fv) * c32[idx] + sigmf(iv) * tanhf(gv);
      const float hn = sigmf(ov) * tanhf(cn);
      c32[idx]   = cn;
      h_out[idx] = (__bf16)hn;
      if (c_bf)  c_bf[idx]  = (__bf16)cn;
      if (h_f32) h_f32[idx] = hn;
      if (h_copy) h_copy[(size_t)m * copy_stride + n] = (__bf16)hn;
    }
  }
}

// Vocab projection fused with log-sum-exp + gold-logit capture.
// Row m = t*64+b ; gold id = tgt_ids[m+64] (tgt_ids[1:]).  |logit| is O(1)
// (0.05-scale weights, tanh inputs) so sum exp(z) needs no max-subtraction.
__global__ void vocab_lse(GemmArgs g, const int* tgt_ids, float* row_sum,
                          float* row_gold) {
  const int wave   = blockIdx.x * (blockDim.x >> 5) + (threadIdx.x >> 5);
  const int tilesM = g.M >> 5;
  const int tilesN = g.N >> 6;
  if (wave >= tilesM * tilesN) return;
  const int tileM = (wave % tilesM) << 5;
  const int tileN = (wave / tilesM) << 6;

  v8f zero = {0.f, 0.f, 0.f, 0.f, 0.f, 0.f, 0.f, 0.f};
  v8f acc[2][4] = {{zero, zero, zero, zero}, {zero, zero, zero, zero}};
  gemm_mainloop(g, tileM, tileN, acc);

  const int lane = threadIdx.x & 31;
  const int half = lane >> 4;
  const int l15  = lane & 15;
#pragma unroll
  for (int mi = 0; mi < 2; ++mi) {
    union { v8f v; float e[8]; } C[4];
#pragma unroll
    for (int j = 0; j < 4; ++j) C[j].v = acc[mi][j];
#pragma unroll
    for (int r = 0; r < 8; ++r) {
      const int m   = tileM + mi * 16 + r + 8 * half;
      const int gid = tgt_ids[m + 64];
      float s = 0.f;
#pragma unroll
      for (int j = 0; j < 4; ++j) {
        const float v = C[j].e[r];
        const int   n = tileN + j * 16 + l15;
        if (n == gid) row_gold[m] = v;   // unique writer per row
        s += __expf(v);
      }
      // reduce the 16 lanes of this half-wave sharing row m
      for (int msk = 8; msk >= 1; msk >>= 1) s += __shfl_xor(s, msk, 32);
      if (l15 == 0) atomicAdd(&row_sum[m], s);
    }
  }
}

// Per-batch attention: e=h.enc_proj, softmax over S, a_t=alpha.enc_hiddens
__global__ void attention(const float* hdec, const float* encP,
                          const __bf16* encH, __bf16* a_t) {
  const int b = blockIdx.x, tid = threadIdx.x;
  const int wv = tid >> 5, lane = tid & 31;
  __shared__ float e[NS];
  for (int j = 0; j < 6; ++j) {                // 8 waves x 6 = 48 positions
    const int s = wv * 6 + j;
    const float* ep = encP + ((size_t)b * NS + s) * NH;
    const float* hp = hdec + (size_t)b * NH;
    float sum = 0.f;
    for (int k = lane; k < NH; k += 32) sum += hp[k] * ep[k];
    for (int m = 16; m >= 1; m >>= 1) sum += __shfl_xor(sum, m, 32);
    if (lane == 0) e[s] = sum;
  }
  __syncthreads();
  if (tid == 0) {
    float mx = -1e30f;
    for (int s = 0; s < NS; ++s) mx = fmaxf(mx, e[s]);
    float ss = 0.f;
    for (int s = 0; s < NS; ++s) { e[s] = expf(e[s] - mx); ss += e[s]; }
    const float inv = 1.f / ss;
    for (int s = 0; s < NS; ++s) e[s] *= inv;
  }
  __syncthreads();
  for (int d = tid; d < 2 * NH; d += 256) {
    float s = 0.f;
    const __bf16* hb = encH + (size_t)b * NS * 2 * NH + d;
    for (int si = 0; si < NS; ++si) s += e[si] * (float)hb[(size_t)si * 2 * NH];
    a_t[(size_t)b * 2 * NH + d] = (__bf16)s;
  }
}

// Gather src embeddings for the width-2 CNN: row (t*64+b) = [X[t]; X[t+1]] bf16
__global__ void build_acnn(const int* src_ids, const float* emb, __bf16* A) {
  const size_t idx = (size_t)blockIdx.x * 256 + threadIdx.x;
  if (idx >= (size_t)NS * NB * 2 * NE) return;
  const int row = (int)(idx >> 10), col = (int)(idx & 1023);
  const int t = row / NB, b = row % NB;
  float v = 0.f;
  if (col < NE)           v = emb[(size_t)src_ids[t * NB + b] * NE + col];
  else if (t + 1 < NS)    v = emb[(size_t)src_ids[(t + 1) * NB + b] * NE + (col - NE)];
  A[idx] = (__bf16)v;
}

__global__ void build_y(const int* tgt_ids, const float* emb, __bf16* Y) {
  const size_t idx = (size_t)blockIdx.x * 256 + threadIdx.x;
  if (idx >= (size_t)(NT - 1) * NB * NE) return;
  const int row = (int)(idx / NE), col = (int)(idx % NE);
  Y[idx] = (__bf16)emb[(size_t)tgt_ids[row] * NE + col];
}

// cnn_w (O,I,2) -> Wcnn (O, 2I) bf16, matching the [X[t]; X[t+1]] K-order
__global__ void pack_wcnn(const float* w, __bf16* Wcnn) {
  const size_t idx = (size_t)blockIdx.x * 256 + threadIdx.x;
  if (idx >= (size_t)NE * 2 * NE) return;
  const int n = (int)(idx >> 10), k = (int)(idx & 1023);
  Wcnn[idx] = (__bf16)((k < NE) ? w[(size_t)n * 2 * NE + k * 2]
                                : w[(size_t)n * 2 * NE + (k - NE) * 2 + 1]);
}

// fp32 -> bf16 weight conversion (vectorized by 4)
__global__ void wcvt(const float* __restrict__ src, __bf16* __restrict__ dst, int n4) {
  const int i = blockIdx.x * 256 + threadIdx.x;
  if (i >= n4) return;
  const float4 v = ((const float4*)src)[i];
  __bf16* d = dst + (size_t)i * 4;
  d[0] = (__bf16)v.x; d[1] = (__bf16)v.y; d[2] = (__bf16)v.z; d[3] = (__bf16)v.w;
}

__global__ void finalize(const float* row_sum, const float* row_gold,
                         const int* tgt_ids, float* out) {
  const int r = blockIdx.x * 256 + threadIdx.x;
  if (r >= (NT - 1) * NB) return;
  const int gid = tgt_ids[r + NB];
  if (gid != 0) {
    const float lp = row_gold[r] - logf(row_sum[r]);
    atomicAdd(&out[r & (NB - 1)], lp);
  }
}

static inline void launch_gemm(const GemmArgs& g, const float* b1, const float* b2,
                               float* oF, __bf16* oB, int act, hipStream_t s) {
  const int waves = (g.M / 32) * (g.N / 64);
  gemm_bias_act<<<(waves + 7) / 8, 256, 0, s>>>(g, b1, b2, oF, oB, act);
}

extern "C" void kernel_launch(void* const* d_in, const int* in_sizes, int n_in,
                              void* d_out, int out_size, void* d_ws, size_t ws_size,
                              hipStream_t stream) {
  const int*   src_ids = (const int*)d_in[0];
  const int*   tgt_ids = (const int*)d_in[1];
  const float* src_emb = (const float*)d_in[3];
  const float* tgt_emb = (const float*)d_in[4];
  const float* cnn_w   = (const float*)d_in[5];
  const float* cnn_b   = (const float*)d_in[6];
  const float* bih_f   = (const float*)d_in[9];
  const float* bhh_f   = (const float*)d_in[10];
  const float* bih_b   = (const float*)d_in[13];
  const float* bhh_b   = (const float*)d_in[14];
  const float* dbih    = (const float*)d_in[17];
  const float* dbhh    = (const float*)d_in[18];
  float* out = (float*)d_out;

  // ---- workspace carve (~180 MB) ----
  char* ws = (char*)d_ws;
  size_t off = 0;
  auto carve = [&](size_t bytes) -> char* {
    char* p = ws + off;
    off += (bytes + 255) & ~(size_t)255;
    return p;
  };
  // activations / state
  __bf16* Acnn = (__bf16*)carve((size_t)NS * NB * 2 * NE * 2);
  __bf16* Wcn  = (__bf16*)carve((size_t)NE * 2 * NE * 2);
  __bf16* Xc   = (__bf16*)carve((size_t)NS * NB * NE * 2);
  __bf16* Yb   = (__bf16*)carve((size_t)(NT - 1) * NB * NE * 2);
  __bf16* encH = (__bf16*)carve((size_t)NB * NS * 2 * NH * 2);
  float*  encP = (float*) carve((size_t)NB * NS * NH * 4);
  __bf16* hf[2] = {(__bf16*)carve((size_t)NB * NH * 2), (__bf16*)carve((size_t)NB * NH * 2)};
  __bf16* hb[2] = {(__bf16*)carve((size_t)NB * NH * 2), (__bf16*)carve((size_t)NB * NH * 2)};
  __bf16* hd[2] = {(__bf16*)carve((size_t)NB * NH * 2), (__bf16*)carve((size_t)NB * NH * 2)};
  float*  cf   = (float*) carve((size_t)NB * NH * 4);
  float*  cb   = (float*) carve((size_t)NB * NH * 4);
  float*  cd   = (float*) carve((size_t)NB * NH * 4);
  float*  hd32 = (float*) carve((size_t)NB * NH * 4);
  __bf16* cfb  = (__bf16*)carve((size_t)NB * NH * 2);
  __bf16* cbb  = (__bf16*)carve((size_t)NB * NH * 2);
  __bf16* at   = (__bf16*)carve((size_t)NB * 2 * NH * 2);
  __bf16* Comb = (__bf16*)carve((size_t)NT * NB * NH * 2);  // block0 = zeros (o_prev@t0)
  float*  rsum = (float*) carve((size_t)(NT - 1) * NB * 4);
  float*  rgld = (float*) carve((size_t)(NT - 1) * NB * 4);
  // bf16 weight copies
  const size_t zWihE = (size_t)4 * NH * NE;        // 4096x512
  const size_t zWhh  = (size_t)4 * NH * NH;        // 4096x1024
  const size_t zdWih = (size_t)4 * NH * (NH + NE); // 4096x1536
  const size_t z2H   = (size_t)NH * 2 * NH;        // 1024x2048
  const size_t z3H   = (size_t)NH * 3 * NH;        // 1024x3072
  const size_t zVoc  = (size_t)NV * NH;            // 32000x1024
  __bf16* Wih_f = (__bf16*)carve(zWihE * 2);
  __bf16* Whh_f = (__bf16*)carve(zWhh  * 2);
  __bf16* Wih_b = (__bf16*)carve(zWihE * 2);
  __bf16* Whh_b = (__bf16*)carve(zWhh  * 2);
  __bf16* dWih  = (__bf16*)carve(zdWih * 2);
  __bf16* dWhh  = (__bf16*)carve(zWhh  * 2);
  __bf16* Wh    = (__bf16*)carve(z2H   * 2);
  __bf16* Wc    = (__bf16*)carve(z2H   * 2);
  __bf16* Watt  = (__bf16*)carve(z2H   * 2);
  __bf16* Wcomb = (__bf16*)carve(z3H   * 2);
  __bf16* Wvoc  = (__bf16*)carve(zVoc  * 2);
  (void)ws_size; (void)in_sizes; (void)n_in; (void)out_size;

  // ---- init ----
  hipMemsetAsync(out,  0, NB * sizeof(float), stream);
  hipMemsetAsync(rsum, 0, (NT - 1) * NB * sizeof(float), stream);
  hipMemsetAsync(Comb, 0, (size_t)NB * NH * 2, stream);
  hipMemsetAsync(hf[0], 0, (size_t)NB * NH * 2, stream);
  hipMemsetAsync(hb[0], 0, (size_t)NB * NH * 2, stream);
  hipMemsetAsync(cf,   0, (size_t)NB * NH * 4, stream);
  hipMemsetAsync(cb,   0, (size_t)NB * NH * 4, stream);

  // weight f32 -> bf16
  auto cvt = [&](const void* src, __bf16* dst, size_t n) {
    const int n4 = (int)(n / 4);
    wcvt<<<(n4 + 255) / 256, 256, 0, stream>>>((const float*)src, dst, n4);
  };
  cvt(d_in[7],  Wih_f, zWihE);
  cvt(d_in[8],  Whh_f, zWhh);
  cvt(d_in[11], Wih_b, zWihE);
  cvt(d_in[12], Whh_b, zWhh);
  cvt(d_in[15], dWih,  zdWih);
  cvt(d_in[16], dWhh,  zWhh);
  cvt(d_in[19], Wh,    z2H);
  cvt(d_in[20], Wc,    z2H);
  cvt(d_in[21], Watt,  z2H);
  cvt(d_in[22], Wcomb, z3H);
  cvt(d_in[23], Wvoc,  zVoc);

  build_acnn<<<(NS * NB * 2 * NE + 255) / 256, 256, 0, stream>>>(src_ids, src_emb, Acnn);
  build_y<<<((NT - 1) * NB * NE + 255) / 256, 256, 0, stream>>>(tgt_ids, tgt_emb, Yb);
  pack_wcnn<<<(NE * 2 * NE + 255) / 256, 256, 0, stream>>>(cnn_w, Wcn);

  // ---- CNN: Xc(3072x512) = Acnn(3072x1024) * Wcnn^T + cnn_b ----
  {
    GemmArgs g = {Acnn, 2 * NE, nullptr, 0, nullptr, 0,
                  Wcn, 2 * NE, nullptr, 0, NS * NB, NE, 2 * NE};
    launch_gemm(g, cnn_b, nullptr, nullptr, Xc, 0, stream);
  }

  // ---- bidirectional LSTM encoder (fused gates+nonlinearity per step) ----
  for (int t = 0; t < NS; ++t) {
    const int pi = t & 1, po = (t + 1) & 1;
    {
      GemmArgs g = {Xc + (size_t)t * NB * NE, NE, hf[pi], NH, nullptr, 0,
                    Wih_f, NE, Whh_f, NH, NB, 4 * NH, NE + NH};
      lstm_step_fused<<<16, 256, 0, stream>>>(g, bih_f, bhh_f, cf, hf[po], cfb,
                                              nullptr, encH + (size_t)t * 2 * NH,
                                              NS * 2 * NH);
    }
    {
      const int tb = NS - 1 - t;
      GemmArgs g = {Xc + (size_t)tb * NB * NE, NE, hb[pi], NH, nullptr, 0,
                    Wih_b, NE, Whh_b, NH, NB, 4 * NH, NE + NH};
      lstm_step_fused<<<16, 256, 0, stream>>>(g, bih_b, bhh_b, cb, hb[po], cbb,
                                              nullptr, encH + (size_t)tb * 2 * NH + NH,
                                              NS * 2 * NH);
    }
  }
  // after 48 steps the final h is back in buffer 0
  __bf16* hfF = hf[NS & 1];
  __bf16* hbF = hb[NS & 1];

  // ---- enc_proj = enc_hiddens @ Watt^T ----
  {
    GemmArgs g = {encH, 2 * NH, nullptr, 0, nullptr, 0,
                  Watt, 2 * NH, nullptr, 0, NB * NS, NH, 2 * NH};
    launch_gemm(g, nullptr, nullptr, encP, nullptr, 0, stream);
  }
  // ---- decoder init: dec_h = [hf,hb]@Wh^T ; dec_c = [cf,cb]@Wc^T ----
  {
    GemmArgs g = {hfF, NH, hbF, NH, nullptr, 0, Wh, 2 * NH, nullptr, 0, NB, NH, 2 * NH};
    launch_gemm(g, nullptr, nullptr, hd32, hd[0], 0, stream);
  }
  {
    GemmArgs g = {cfb, NH, cbb, NH, nullptr, 0, Wc, 2 * NH, nullptr, 0, NB, NH, 2 * NH};
    launch_gemm(g, nullptr, nullptr, cd, nullptr, 0, stream);
  }

  // ---- attention decoder ----
  for (int t = 0; t < NT - 1; ++t) {
    const int pi = t & 1, po = (t + 1) & 1;
    GemmArgs g = {Comb + (size_t)t * NB * NH, NH,        // o_prev
                  Yb + (size_t)t * NB * NE, NE,          // Y_t
                  hd[pi], NH,                            // h
                  dWih, NH + NE, dWhh, NH,
                  NB, 4 * NH, 2 * NH + NE + NH};
    lstm_step_fused<<<16, 256, 0, stream>>>(g, dbih, dbhh, cd, hd[po], nullptr,
                                            hd32, nullptr, 0);
    attention<<<NB, 256, 0, stream>>>(hd32, encP, encH, at);
    GemmArgs gc = {at, 2 * NH, hd[po], NH, nullptr, 0,
                   Wcomb, 3 * NH, nullptr, 0, NB, NH, 3 * NH};
    launch_gemm(gc, nullptr, nullptr, nullptr, Comb + (size_t)(t + 1) * NB * NH, 1, stream);
  }

  // ---- vocab projection + fused log-sum-exp, then loss accumulation ----
  {
    GemmArgs g = {Comb + (size_t)NB * NH, NH, nullptr, 0, nullptr, 0,
                  Wvoc, NH, nullptr, 0, (NT - 1) * NB, NV, NH};
    const int waves = (g.M / 32) * (g.N / 64);
    vocab_lse<<<(waves + 7) / 8, 256, 0, stream>>>(g, tgt_ids, rsum, rgld);
  }
  finalize<<<((NT - 1) * NB + 255) / 256, 256, 0, stream>>>(rsum, rgld, tgt_ids, out);
}
